// AIS_Estimator_61186104099351
// MI455X (gfx1250) — matrix operand order repaired and checked by reading, not compiled, not run on hardware
//
#include <hip/hip_runtime.h>
#include <stdint.h>

// Problem constants (fixed by the reference)
#define NSMP   1024
#define BB     128
#define DIMN   64
#define KSTEPS 16
#define NLEAP  3
#define STEPSZ 0.05f
#define LOG2PI 1.8378770664093453f

typedef __bf16 v16bf __attribute__((ext_vector_type(16)));
typedef __bf16 v2bf  __attribute__((ext_vector_type(2)));
typedef float  v8f   __attribute__((ext_vector_type(8)));

union FragU { v16bf v; int4 q[2]; };

// Annealing schedule beta_i = (sigmoid(8i/17-4)-s0)/(s17-s0), precomputed.
__constant__ float BETA_C[KSTEPS + 2] = {
    0.0f,       0.0108924f, 0.0278538f, 0.0538505f, 0.0927437f, 0.1488763f,
    0.2258245f, 0.3241848f, 0.4392618f, 0.5607382f, 0.6758152f, 0.7741755f,
    0.8511237f, 0.9072563f, 0.9461495f, 0.9721462f, 0.9891076f, 1.0f};

// ---------------- stateless RNG (momentum refresh) ----------------
__device__ __forceinline__ uint32_t hmix(uint32_t x) {
  x ^= x >> 16; x *= 0x7feb352dU;
  x ^= x >> 15; x *= 0x846ca68bU;
  x ^= x >> 16; return x;
}
// Two independent N(0,1) draws from one counter (full Box-Muller pair).
__device__ __forceinline__ void rng_normal2(uint32_t s, float& z0, float& z1) {
  uint32_t h1 = hmix(s);
  uint32_t h2 = hmix(s ^ 0x9E3779B9U);
  float u1 = (float)((h1 >> 8) + 1u) * (1.0f / 16777217.0f);  // (0,1]
  float u2 = (float)(h2 >> 8) * (1.0f / 16777216.0f);
  float r = sqrtf(-2.0f * __logf(u1));
  float sn, cs;
  __sincosf(6.28318530718f * u2, &sn, &cs);
  z0 = r * cs;
  z1 = r * sn;
}

// ---------------- WMMA helpers ----------------
#define WMMA_BF16(C, A, B_) \
  (C) = __builtin_amdgcn_wmma_f32_16x16x32_bf16(false, (A), false, (B_), (short)0, (C), false, false)

// Write 16x64 f32 tile (C-layout across lanes) to wave-private LDS strip as
// row-major bf16 (packed converts: low half -> b16 store, high half ->
// ds_store_b16_d16_hi pattern), then read back the two 16x32 A fragments.
// DS ops are in-order per wave; asm memory clobbers fence store->load.
__device__ __forceinline__ void bounce_frags(const v8f (&src)[4], __bf16* bnc,
                                             int lh, int sel,
                                             v16bf& a0, v16bf& a1) {
  asm volatile("" ::: "memory");
  uint16_t* b16p = (uint16_t*)bnc;
#pragma unroll
  for (int v = 0; v < 8; ++v)
#pragma unroll
    for (int t2 = 0; t2 < 2; ++t2) {
      v2bf pk = {(__bf16)src[2 * t2][v], (__bf16)src[2 * t2 + 1][v]};
      uint32_t u = __builtin_bit_cast(uint32_t, pk);
      int idx = (v + 8 * sel) * 64 + t2 * 32 + lh;
      b16p[idx] = (uint16_t)u;            // element (2*t2,   v)
      b16p[idx + 16] = (uint16_t)(u >> 16);  // element (2*t2+1, v)
    }
  asm volatile("s_wait_dscnt 0" ::: "memory");
  // A layout: lanes<16 take K kb+{0..7,16..23}, lanes>=16 take kb+{8..15,24..31}
  FragU f0, f1;
  int base = lh * 64 + sel * 8;
  f0.q[0] = *(const int4*)(bnc + base);
  f0.q[1] = *(const int4*)(bnc + base + 16);
  f1.q[0] = *(const int4*)(bnc + base + 32);
  f1.q[1] = *(const int4*)(bnc + base + 48);
  a0 = f0.v;
  a1 = f1.v;
}

// out(16x64) = A(16x64) @ W(64x64): 8 WMMA against pinned B fragments.
__device__ __forceinline__ void mm8(const v16bf a0, const v16bf a1,
                                    const v16bf (&Bf)[2][4], v8f (&out)[4]) {
#pragma unroll
  for (int nt = 0; nt < 4; ++nt) {
    v8f c;
#pragma unroll
    for (int i = 0; i < 8; ++i) c[i] = 0.0f;
    WMMA_BF16(c, a0, Bf[0][nt]);
    WMMA_BF16(c, a1, Bf[1][nt]);
    out[nt] = c;
  }
}

// ---------------- prep kernels (tiny) ----------------
// M = W_dec @ W_dec^T  (64x64, symmetric)
__global__ void prep_M(const float* __restrict__ Wd, float* __restrict__ M) {
  int d = blockIdx.x, j = threadIdx.x;
  float acc = 0.0f;
#pragma unroll 8
  for (int f = 0; f < DIMN; ++f) acc += Wd[d * DIMN + f] * Wd[j * DIMN + f];
  M[d * DIMN + j] = acc;
}
// mu = x @ W_enc ; c = x @ W_dec^T   (both 128x64)
__global__ void prep_muc(const float* __restrict__ x, const float* __restrict__ W_enc,
                         const float* __restrict__ Wd, float* __restrict__ mu,
                         float* __restrict__ c) {
  int b = blockIdx.x, j = threadIdx.x;
  float a1 = 0.0f, a2 = 0.0f;
#pragma unroll 8
  for (int k = 0; k < DIMN; ++k) {
    float xv = x[b * DIMN + k];
    a1 += xv * W_enc[k * DIMN + j];
    a2 += xv * Wd[j * DIMN + k];
  }
  mu[b * DIMN + j] = a1;
  c[b * DIMN + j] = a2;
}

// ---------------- main kernel: full AIS chains, state in registers ----------------
__global__ __launch_bounds__(256) void ais_chains(
    const float* __restrict__ x, const float* __restrict__ Wd,
    const float* __restrict__ qn, const float* __restrict__ mu,
    const float* __restrict__ cvec, const float* __restrict__ Mg,
    float* __restrict__ slw_out) {
  __shared__ __align__(16) __bf16 sM[64 * 64];     // M symmetric => [N][K] image is M
  __shared__ __align__(16) __bf16 sWdT[64 * 64];   // [e][d] : B for r = q@Wdec
  __shared__ __align__(16) __bf16 sB[8][16 * 64];  // per-wave bounce strips

  const int tid = threadIdx.x;
  for (int i = tid; i < 64 * 64; i += 256) {
    int d = i >> 6, e = i & 63;
    sM[i] = (__bf16)Mg[i];
    sWdT[e * 64 + d] = (__bf16)Wd[i];
  }
  __syncthreads();

  const int wave = tid >> 5, lane = tid & 31;
  const int lh = lane & 15, sel = lane >> 4;
  const int gwave = blockIdx.x * 8 + wave;
  const int b = gwave >> 6;              // 64 waves per b
  const int nbase = (gwave & 63) << 4;   // 16 chains per wave
  __bf16* bnc = &sB[wave][0];

  // Pin B fragments for M (gradient GEMM) and Wdec (residual GEMM) in VGPRs.
  v16bf BM[2][4], BW[2][4];
#pragma unroll
  for (int kb = 0; kb < 2; ++kb)
#pragma unroll
    for (int nt = 0; nt < 4; ++nt) {
      int base = (nt * 16 + lh) * 64 + kb * 32 + sel * 8;
      FragU f;
      f.q[0] = *(const int4*)(sM + base);
      f.q[1] = *(const int4*)(sM + base + 16);
      BM[kb][nt] = f.v;
      FragU g;
      g.q[0] = *(const int4*)(sWdT + base);
      g.q[1] = *(const int4*)(sWdT + base + 16);
      BW[kb][nt] = g.v;
    }

  float x_l[4], mu_l[4], nmu[4], dcm[4];
  float summu2 = 0.0f;
#pragma unroll
  for (int t = 0; t < 4; ++t) {
    x_l[t] = x[b * DIMN + t * 16 + lh];
    mu_l[t] = mu[b * DIMN + t * 16 + lh];
    dcm[t] = cvec[b * DIMN + t * 16 + lh] - mu_l[t];  // c - mu
    nmu[t] = -mu_l[t];
    summu2 += mu_l[t] * mu_l[t];
  }

  // q0 = mu + q_noise, loaded straight into C-layout registers.
  v8f q[4];
#pragma unroll
  for (int t = 0; t < 4; ++t)
#pragma unroll
    for (int v = 0; v < 8; ++v) {
      int n = nbase + v + 8 * sel;
      q[t][v] = mu_l[t] + qn[((size_t)(n * BB + b)) * DIMN + t * 16 + lh];
    }

  // Loop-invariant RNG seed bases (only the k-dependent XOR changes per step).
  uint32_t seedb[2][8];
#pragma unroll
  for (int t2 = 0; t2 < 2; ++t2)
#pragma unroll
    for (int v = 0; v < 8; ++v) {
      int n = nbase + v + 8 * sel;
      seedb[t2][v] = (uint32_t)(((n * BB + b) << 6) + t2 * 32 + lh);
    }

  float slwp[8];
#pragma unroll
  for (int v = 0; v < 8; ++v) slwp[v] = 0.0f;

  v8f gm[4], r[4], p[4];
  v16bf a0, a1;
  // Initial eval at q0: one bounce feeds both gm = q0@M and r = q0@Wdec.
  bounce_frags(q, bnc, lh, sel, a0, a1);
  mm8(a0, a1, BM, gm);
  mm8(a0, a1, BW, r);

  {  // slw0 = beta1 * w(q0); per-elem contrib = -e*e/2 - q*mu
     // (the +mu^2/2 and -32*log2pi pieces telescope: sum_k coef == 1)
    float coef = BETA_C[1];
#pragma unroll
    for (int v = 0; v < 8; ++v) {
      float acc = 0.0f;
#pragma unroll
      for (int t = 0; t < 4; ++t) {
        float e = x_l[t] - r[t][v];
        acc = fmaf(e * e, -0.5f, acc);
        acc = fmaf(q[t][v], nmu[t], acc);
      }
      slwp[v] = fmaf(coef, acc, slwp[v]);
    }
  }

#pragma unroll 1
  for (int k = 1; k <= KSTEPS; ++k) {
    float bk = BETA_C[k];
    float coef = BETA_C[k + 1] - bk;
    float nbk = -bk;
    uint32_t kxor = 0x9E3779B9u * (uint32_t)k;  // uniform -> SALU
    float m2[4];
#pragma unroll
    for (int t = 0; t < 4; ++t) m2[t] = fmaf(bk, dcm[t], mu_l[t]);  // bk*c+(1-bk)*mu

    // full momentum resample: one Box-Muller pair fills elements t and t+1
    // (counter-based hash; JAX threefry not bit-reproducible, no runtime check)
#pragma unroll
    for (int t2 = 0; t2 < 2; ++t2)
#pragma unroll
      for (int v = 0; v < 8; ++v) {
        float z0, z1;
        rng_normal2(seedb[t2][v] ^ kxor, z0, z1);
        p[2 * t2][v] = z0;
        p[2 * t2 + 1][v] = z1;
      }

    // leading half-kick reuses gm cached from previous step's last eval:
    // g = bk*(c - q@M) + (1-bk)*mu - q = m2 - bk*gm - q
#pragma unroll
    for (int t = 0; t < 4; ++t)
#pragma unroll
      for (int v = 0; v < 8; ++v) {
        float g = fmaf(nbk, gm[t][v], m2[t]) - q[t][v];
        p[t][v] = fmaf(0.5f * STEPSZ, g, p[t][v]);
      }

#pragma unroll 1
    for (int l = 0; l < NLEAP; ++l) {
#pragma unroll
      for (int t = 0; t < 4; ++t)
#pragma unroll
        for (int v = 0; v < 8; ++v) q[t][v] = fmaf(STEPSZ, p[t][v], q[t][v]);
      bounce_frags(q, bnc, lh, sel, a0, a1);
      mm8(a0, a1, BM, gm);
      if (l == NLEAP - 1) mm8(a0, a1, BW, r);  // residual shares the bounce
      float sc = (l == NLEAP - 1) ? 0.5f * STEPSZ : STEPSZ;  // fused half-kicks
#pragma unroll
      for (int t = 0; t < 4; ++t)
#pragma unroll
        for (int v = 0; v < 8; ++v) {
          float g = fmaf(nbk, gm[t][v], m2[t]) - q[t][v];
          p[t][v] = fmaf(sc, g, p[t][v]);
        }
    }

    // incremental weight at the final q of this step (r at final q)
#pragma unroll
    for (int v = 0; v < 8; ++v) {
      float acc = 0.0f;
#pragma unroll
      for (int t = 0; t < 4; ++t) {
        float e = x_l[t] - r[t][v];
        acc = fmaf(e * e, -0.5f, acc);
        acc = fmaf(q[t][v], nmu[t], acc);
      }
      slwp[v] = fmaf(coef, acc, slwp[v]);
    }
  }

  // Row reduction across the 16 lanes of each half.  The telescoped constants
  // (sum_k coef == 1) contribute  +mu^2/2  and  -32*log(2pi)  per row.
  const float cfinal = 0.5f * summu2 - 0.5f * (float)DIMN * LOG2PI;
#pragma unroll
  for (int v = 0; v < 8; ++v) {
    float s = slwp[v] + cfinal;
    s += __shfl_xor(s, 1, 16);
    s += __shfl_xor(s, 2, 16);
    s += __shfl_xor(s, 4, 16);
    s += __shfl_xor(s, 8, 16);
    if (lh == 0) slw_out[b * NSMP + nbase + v + 8 * sel] = s;
  }
}

// ---------------- final kernel: per-b logsumexp over n ----------------
__global__ void lse_kernel(const float* __restrict__ slw, float* __restrict__ out) {
  __shared__ float red[256];
  int b = blockIdx.x, tid = threadIdx.x;
  float m = -3.4e38f;
  for (int n = tid; n < NSMP; n += 256) m = fmaxf(m, slw[b * NSMP + n]);
  red[tid] = m;
  __syncthreads();
  for (int s = 128; s > 0; s >>= 1) {
    if (tid < s) red[tid] = fmaxf(red[tid], red[tid + s]);
    __syncthreads();
  }
  m = red[0];
  __syncthreads();
  float sum = 0.0f;
  for (int n = tid; n < NSMP; n += 256) sum += __expf(slw[b * NSMP + n] - m);
  red[tid] = sum;
  __syncthreads();
  for (int s = 128; s > 0; s >>= 1) {
    if (tid < s) red[tid] += red[tid + s];
    __syncthreads();
  }
  if (tid == 0) out[b] = m + __logf(red[0]) - __logf((float)NSMP);
}

extern "C" void kernel_launch(void* const* d_in, const int* in_sizes, int n_in,
                              void* d_out, int out_size, void* d_ws, size_t ws_size,
                              hipStream_t stream) {
  (void)in_sizes; (void)n_in; (void)out_size; (void)ws_size;
  const float* x     = (const float*)d_in[0];
  const float* W_enc = (const float*)d_in[1];
  const float* W_dec = (const float*)d_in[2];
  const float* q_noi = (const float*)d_in[3];
  // d_in[4] (p_noise) is dead: p is resampled before first use in the scan.

  float* mu  = (float*)d_ws;             // 128*64 f32
  float* c   = mu + BB * DIMN;           // 128*64 f32
  float* M   = c + BB * DIMN;            // 64*64 f32
  float* slw = M + DIMN * DIMN;          // 128*1024 f32

  prep_M<<<DIMN, DIMN, 0, stream>>>(W_dec, M);
  prep_muc<<<BB, DIMN, 0, stream>>>(x, W_enc, W_dec, mu, c);
  ais_chains<<<(NSMP / 16) * BB / 8, 256, 0, stream>>>(x, W_dec, q_noi, mu, c, M, slw);
  lse_kernel<<<BB, 256, 0, stream>>>(slw, (float*)d_out);
}
